// CBR_RNN_45208825758361
// MI455X (gfx1250) — matrix operand order repaired and checked by reading, not compiled, not run on hardware
//
#include <hip/hip_runtime.h>

// ---------------- problem constants ----------------
#define NWG     16          // persistent workgroups (co-resident; MI455X has far more WGPs)
#define BSZ     16
#define SEQLEN  128
#define TCACHE  129
#define NHID    512
#define NINP    512
#define NHEADS  8
#define DHEAD   64
#define NTOK    32001
#define NDECP   32016       // 32001 padded to multiple of 16 (zero-padded weights)

typedef __attribute__((ext_vector_type(16))) __bf16        v16bf;
typedef __attribute__((ext_vector_type(8)))  float         v8f;
typedef __attribute__((ext_vector_type(4)))  unsigned int  v4u;

union AFrag { v16bf bf; v4u u[2]; };

__device__ __forceinline__ unsigned short f2bf(float f) {
  union { float f; unsigned u; } v; v.f = f;
  unsigned u = v.u;
  return (unsigned short)((u + 0x7fffu + ((u >> 16) & 1u)) >> 16);  // RNE truncate
}

// ---------------- wave-level WMMA GEMM tile (M=16, N=16, K loop) ----------------
// A is bf16 activations, rows 0..15, possibly split into 512-wide K segments
// (zero-copy concat). B is a bf16 weight matrix stored [N_total][K] row-major,
// which is exactly the K-major-per-column layout the B fragment wants.
struct Aseg { const unsigned short* p; int stride; };

__device__ __forceinline__ v8f wave_gemm(const Aseg* segs,
                                         const unsigned short* __restrict__ W,
                                         int ldw, int n0, int K, int lane) {
  const int m    = lane & 15;   // A row index == B column index (per ISA layouts)
  const int half = lane >> 4;
  v8f acc = {0.f,0.f,0.f,0.f,0.f,0.f,0.f,0.f};
  for (int kc = 0; kc < K; kc += 32) {
    // A 16x32 bf16 fragment: lane half=0 owns K {kc..kc+7, kc+16..kc+23},
    // half=1 owns K {kc+8..kc+15, kc+24..kc+31}  -> two 16B loads.
    const int k0 = kc + 8 * half;
    const Aseg sg = segs[k0 >> 9];
    const unsigned short* pa = sg.p + (size_t)m * sg.stride + (k0 & 511);
    AFrag a, b;
    a.u[0] = *(const v4u*)pa;
    a.u[1] = *(const v4u*)(pa + 16);
    // B 32x16 bf16 fragment: lane owns column n=n0+m, K {kc+8*half..+7, kc+16+8*half..+7} (same as A).
    const unsigned short* pb = W + (size_t)(n0 + m) * ldw + kc + 8 * half;
    b.u[0] = *(const v4u*)pb;
    b.u[1] = *(const v4u*)(pb + 16);
    acc = __builtin_amdgcn_wmma_f32_16x16x32_bf16(false, a.bf, false, b.bf,
                                                  (short)0, acc, false, false);
  }
  return acc;
}

// C/D layout: lane -> n = n0 + (lane&15); VGPR r -> m = r + 8*(lane>>4)
__device__ __forceinline__ void store_f32_bias(v8f acc, float* dst, int ldd, int n0,
                                               const float* __restrict__ bias, int lane) {
  const int n = n0 + (lane & 15);
  const int mh = (lane >> 4) * 8;
  const float bv = bias[n];
#pragma unroll
  for (int r = 0; r < 8; ++r) dst[(size_t)(mh + r) * ldd + n] = acc[r] + bv;
}

__device__ __forceinline__ void store_bf16_bias(v8f acc, unsigned short* dst, int ldd, int n0,
                                                const float* __restrict__ bias, int lane) {
  const int n = n0 + (lane & 15);
  const int mh = (lane >> 4) * 8;
  const float bv = bias[n];
#pragma unroll
  for (int r = 0; r < 8; ++r) dst[(size_t)(mh + r) * ldd + n] = f2bf(acc[r] + bv);
}

// ---------------- global barrier over NWG workgroups ----------------
// Fresh counter slot per barrier (init-kernel zeroes the array every launch),
// so there is no reuse hazard and the kernel is deterministic.
__device__ __forceinline__ void global_bar(int* bars, int idx) {
  __syncthreads();
  if (threadIdx.x == 0) {
    __threadfence();
    atomicAdd(&bars[idx], 1);
    volatile int* p = bars + idx;
    while (*p < NWG) { __builtin_amdgcn_s_sleep(2); }
    __threadfence();
  }
  __syncthreads();
}

// ---------------- row LayerNorm + tanh, one wave per row ----------------
__device__ __forceinline__ void ln_tanh_row(const float* __restrict__ raw, int N,
                                            const float* __restrict__ g,
                                            const float* __restrict__ be,
                                            unsigned short* __restrict__ dst,
                                            int lane, int b) {
  float s = 0.f, s2 = 0.f;
  for (int c = lane; c < N; c += 32) { float x = raw[(size_t)b*N + c]; s += x; s2 += x*x; }
  for (int o = 16; o; o >>= 1) { s += __shfl_xor(s, o, 32); s2 += __shfl_xor(s2, o, 32); }
  const float mean = s / (float)N;
  const float var  = s2 / (float)N - mean * mean;
  const float inv  = rsqrtf(var + 1e-5f);
  for (int c = lane; c < N; c += 32) {
    float y = tanhf((raw[(size_t)b*N + c] - mean) * inv * g[c] + be[c]);
    dst[(size_t)b*N + c] = f2bf(y);
  }
}

// ---------------- prep kernels ----------------
__global__ void cvt_bf16_kernel(const float* __restrict__ src, unsigned short* __restrict__ dst,
                                long n, long nsrc) {
  long i = (long)blockIdx.x * blockDim.x + threadIdx.x;
  const long stride = (long)gridDim.x * blockDim.x;
  for (; i < n; i += stride) dst[i] = (i < nsrc) ? f2bf(src[i]) : (unsigned short)0;
}

__global__ void emb_kernel(const int* __restrict__ obs, const float* __restrict__ tbl,
                           unsigned short* __restrict__ e16) {
  long i = (long)blockIdx.x * blockDim.x + threadIdx.x;
  const long n = (long)SEQLEN * BSZ * NINP;
  const long stride = (long)gridDim.x * blockDim.x;
  for (; i < n; i += stride) {
    const int c = (int)(i % NINP);
    const long r = i / NINP;
    const int b = (int)(r % BSZ);
    const int s = (int)(r / BSZ);
    const int tok = obs[b * SEQLEN + s];
    e16[i] = f2bf(tbl[(size_t)tok * NINP + c]);   // E16[s][b][c]
  }
}

__global__ void init_kernel(int* __restrict__ bars, unsigned short* __restrict__ h16,
                            float* __restrict__ kp, float* __restrict__ vp,
                            const float* __restrict__ ipb) {
  const int i = blockIdx.x * blockDim.x + threadIdx.x;   // 8192 threads
  if (i < 2048) bars[i] = 0;
  if (i < BSZ * NHID) {
    h16[i] = 0;                                  // h0 = 0 (bf16 zero)
    const int b = i >> 9, c = i & 511;
    // cache row 0 projects the all-zero K0/V0 row -> just the bias
    kp[((size_t)b * TCACHE) * NHID + c] = ipb[512 + c];   // bk
    vp[((size_t)b * TCACHE) * NHID + c] = ipb[1024 + c];  // bv
  }
}

// ---------------- persistent recurrent kernel ----------------
struct RecParams {
  const unsigned short *Wq_w, *Wip, *Wout, *Wint, *Wfin, *E16;
  unsigned short *Q16, *ATNR16, *ATNO16, *H16, *INT16, *FIN16, *HID16;
  float *QLIN, *QP, *INTRAW, *FINRAW, *KP, *VP;
  int *bars;
  const float *q_b, *qn_g, *qn_b, *ipb, *out_b, *int_b, *intn_g, *intn_b,
              *fin_b, *fn_g, *fn_b;
};

__global__ __launch_bounds__(256) void recurrent_kernel(RecParams P) {
  const int lane = threadIdx.x & 31;
  const int wid  = threadIdx.x >> 5;
  const int gw   = blockIdx.x * 8 + wid;          // 0..127 global wave id
  __shared__ float lw[8][132];                    // softmax weights, per wave

  for (int s = 0; s < SEQLEN; ++s) {
    int bi = s * 10;
    const unsigned short* Es = P.E16 + (size_t)s * BSZ * NINP;

    // Phase A: qlin = [e,h] @ q_w.T + q_b   (32 tiles, K=1024)
    if (gw < 32) {
      Aseg segs[2] = { {Es, NINP}, {P.H16, NHID} };
      v8f acc = wave_gemm(segs, P.Wq_w, NINP + NHID, gw * 16, NINP + NHID, lane);
      store_f32_bias(acc, P.QLIN, NHID, gw * 16, P.q_b, lane);
    }
    global_bar(P.bars, bi++);

    // Phase B: q = tanh(LN(qlin))
    if (gw < BSZ) ln_tanh_row(P.QLIN, NHID, P.qn_g, P.qn_b, P.Q16, lane, gw);
    global_bar(P.bars, bi++);

    // Phase C: Qp = q @ Wq.T + bq   (32 tiles, K=512)
    if (gw < 32) {
      Aseg segs[1] = { {P.Q16, NHID} };
      v8f acc = wave_gemm(segs, P.Wip, NHID, gw * 16, NHID, lane);
      store_f32_bias(acc, P.QP, NHID, gw * 16, P.ipb, lane);
    }
    global_bar(P.bars, bi++);

    // Phase D: attention over cached projected Kp/Vp rows 0..s  (128 (b,h) waves)
    {
      const int b = gw >> 3, h = gw & 7;
      const float* qv = P.QP + b * NHID + h * DHEAD;
      const int nt = s + 1;
      float sv[4]; float smax = -3.0e38f; int cnt = 0;
      for (int t = lane; t < nt; t += 32) {
        const float* kv = P.KP + ((size_t)b * TCACHE + t) * NHID + h * DHEAD;
        float d = 0.f;
#pragma unroll
        for (int j = 0; j < DHEAD; ++j) d += qv[j] * kv[j];
        d *= 0.125f;                                   // 1/sqrt(64)
        sv[cnt++] = d; smax = fmaxf(smax, d);
      }
      for (int o = 16; o; o >>= 1) smax = fmaxf(smax, __shfl_xor(smax, o, 32));
      float ssum = 0.f; cnt = 0;
      for (int t = lane; t < nt; t += 32) { float e = __expf(sv[cnt] - smax); sv[cnt] = e; ssum += e; ++cnt; }
      for (int o = 16; o; o >>= 1) ssum += __shfl_xor(ssum, o, 32);
      const float rs = 1.0f / ssum; cnt = 0;
      for (int t = lane; t < nt; t += 32) lw[wid][t] = sv[cnt++] * rs;
      __builtin_amdgcn_wave_barrier();
      float a0 = 0.f, a1 = 0.f;                         // lane owns d=lane, d=lane+32
      for (int t = 0; t < nt; ++t) {
        const float w = lw[wid][t];
        const float* vv = P.VP + ((size_t)b * TCACHE + t) * NHID + h * DHEAD;
        a0 += w * vv[lane]; a1 += w * vv[lane + 32];
      }
      P.ATNR16[b * NHID + h * DHEAD + lane]      = f2bf(a0);
      P.ATNR16[b * NHID + h * DHEAD + lane + 32] = f2bf(a1);
    }
    global_bar(P.bars, bi++);

    // Phase E: attn = attn_raw @ out_w.T + out_b -> bf16  (32 tiles)
    if (gw < 32) {
      Aseg segs[1] = { {P.ATNR16, NHID} };
      v8f acc = wave_gemm(segs, P.Wout, NHID, gw * 16, NHID, lane);
      store_bf16_bias(acc, P.ATNO16, NHID, gw * 16, P.out_b, lane);
    }
    global_bar(P.bars, bi++);

    // Phase F: inter_raw = [e,q,attn,h] @ int_w.T + int_b  (128 tiles, K=2048)
    {
      Aseg segs[4] = { {Es, NINP}, {P.Q16, NHID}, {P.ATNO16, NHID}, {P.H16, NHID} };
      v8f acc = wave_gemm(segs, P.Wint, 2048, gw * 16, 2048, lane);
      store_f32_bias(acc, P.INTRAW, 2048, gw * 16, P.int_b, lane);
    }
    global_bar(P.bars, bi++);

    // Phase G: inter = tanh(LN(inter_raw))
    if (gw < BSZ) ln_tanh_row(P.INTRAW, 2048, P.intn_g, P.intn_b, P.INT16, lane, gw);
    global_bar(P.bars, bi++);

    // Phase H: fin_raw = inter @ fin_w.T + fin_b  (96 tiles, K=2048)
    if (gw < 96) {
      Aseg segs[4] = { {P.INT16, 2048}, {P.INT16 + 512, 2048},
                       {P.INT16 + 1024, 2048}, {P.INT16 + 1536, 2048} };
      v8f acc = wave_gemm(segs, P.Wfin, 2048, gw * 16, 2048, lane);
      store_f32_bias(acc, P.FINRAW, 1536, gw * 16, P.fin_b, lane);
    }
    global_bar(P.bars, bi++);

    // Phase I: final = tanh(LN(fin_raw)); split -> k,v (stay in FIN16), h -> H16 + HID16[s]
    if (gw < BSZ) {
      const int b = gw;
      float su = 0.f, s2 = 0.f;
      for (int c = lane; c < 1536; c += 32) { float x = P.FINRAW[(size_t)b*1536 + c]; su += x; s2 += x*x; }
      for (int o = 16; o; o >>= 1) { su += __shfl_xor(su, o, 32); s2 += __shfl_xor(s2, o, 32); }
      const float mean = su * (1.f / 1536.f);
      const float var  = s2 * (1.f / 1536.f) - mean * mean;
      const float inv  = rsqrtf(var + 1e-5f);
      for (int c = lane; c < 1536; c += 32) {
        const float y = tanhf((P.FINRAW[(size_t)b*1536 + c] - mean) * inv * P.fn_g[c] + P.fn_b[c]);
        const unsigned short hb = f2bf(y);
        P.FIN16[(size_t)b*1536 + c] = hb;
        if (c >= 1024) {                                // h_i slice
          P.H16[b * NHID + (c - 1024)] = hb;
          P.HID16[((size_t)s * BSZ + b) * NHID + (c - 1024)] = hb;
        }
      }
    }
    global_bar(P.bars, bi++);

    // Phase J: incrementally project new cache row: Kp/Vp[:, s+1] = k/v @ Wk/Wv.T + b
    if (gw < 64) {
      const int isV = gw >> 5;
      const int n0  = (gw & 31) * 16;
      Aseg segs[1] = { { P.FIN16 + (isV ? 512 : 0), 1536 } };
      const unsigned short* W = P.Wip + (size_t)(512 + isV * 512) * NHID;
      const float* bias = P.ipb + 512 + isV * 512;
      v8f acc = wave_gemm(segs, W, NHID, n0, NHID, lane);
      float* dst = isV ? P.VP : P.KP;
      const int n = n0 + (lane & 15), mh = (lane >> 4) * 8;
      const float bv = bias[n];
#pragma unroll
      for (int r = 0; r < 8; ++r) {
        const int b = mh + r;
        dst[((size_t)b * TCACHE + (s + 1)) * NHID + n] = acc[r] + bv;
      }
    }
    global_bar(P.bars, bi++);
  }
}

// ---------------- decoder: [2048,512] @ dec_w.T -> [2048,32001]  (store-bound) ----------------
__global__ __launch_bounds__(256) void decoder_kernel(const unsigned short* __restrict__ hid,
                                                      const unsigned short* __restrict__ Wdec,
                                                      const float* __restrict__ dec_b,
                                                      float* __restrict__ out) {
  const int lane = threadIdx.x & 31, wid = threadIdx.x >> 5;
  const int ntile = blockIdx.x * 8 + wid;
  if (ntile >= NDECP / 16) return;
  const int n0 = ntile * 16;
  const int m0 = blockIdx.y * 16;
  Aseg segs[1] = { { hid + (size_t)m0 * NHID, NHID } };
  v8f acc = wave_gemm(segs, Wdec, NHID, n0, NHID, lane);
  const int n = n0 + (lane & 15), mh = (lane >> 4) * 8;
  if (n < NTOK) {
    const float bv = dec_b[n];
#pragma unroll
    for (int r = 0; r < 8; ++r)
      out[(size_t)(m0 + mh + r) * NTOK + n] = acc[r] + bv;
  }
}

// ---------------- host launch ----------------
extern "C" void kernel_launch(void* const* d_in, const int* in_sizes, int n_in,
                              void* d_out, int out_size, void* d_ws, size_t ws_size,
                              hipStream_t stream) {
  (void)in_sizes; (void)n_in; (void)out_size; (void)ws_size;
  const int*   obs    = (const int*)  d_in[0];
  const float* embt   = (const float*)d_in[1];
  const float* q_w    = (const float*)d_in[2];
  const float* q_b    = (const float*)d_in[3];
  const float* qn_g   = (const float*)d_in[4];
  const float* qn_b   = (const float*)d_in[5];
  const float* ip_w   = (const float*)d_in[6];
  const float* ip_b   = (const float*)d_in[7];
  const float* out_w  = (const float*)d_in[8];
  const float* out_b  = (const float*)d_in[9];
  const float* int_w  = (const float*)d_in[10];
  const float* int_b  = (const float*)d_in[11];
  const float* intn_g = (const float*)d_in[12];
  const float* intn_b = (const float*)d_in[13];
  const float* fin_w  = (const float*)d_in[14];
  const float* fin_b  = (const float*)d_in[15];
  const float* fn_g   = (const float*)d_in[16];
  const float* fn_b   = (const float*)d_in[17];
  const float* dec_w  = (const float*)d_in[18];
  const float* dec_b  = (const float*)d_in[19];

  // workspace layout (~64 MB, 256B aligned slices)
  char* ws = (char*)d_ws;
  size_t off = 0;
  auto take = [&](size_t bytes) { size_t o = off; off += (bytes + 255) & ~(size_t)255; return o; };
  unsigned short* Wq16   = (unsigned short*)(ws + take((size_t)512 * 1024 * 2));
  unsigned short* Wip16  = (unsigned short*)(ws + take((size_t)1536 * 512 * 2));
  unsigned short* Wout16 = (unsigned short*)(ws + take((size_t)512 * 512 * 2));
  unsigned short* Wint16 = (unsigned short*)(ws + take((size_t)2048 * 2048 * 2));
  unsigned short* Wfin16 = (unsigned short*)(ws + take((size_t)1536 * 2048 * 2));
  unsigned short* Wdec16 = (unsigned short*)(ws + take((size_t)NDECP * 512 * 2));
  unsigned short* E16    = (unsigned short*)(ws + take((size_t)SEQLEN * BSZ * NINP * 2));
  unsigned short* Q16    = (unsigned short*)(ws + take((size_t)BSZ * NHID * 2));
  unsigned short* ATNR16 = (unsigned short*)(ws + take((size_t)BSZ * NHID * 2));
  unsigned short* ATNO16 = (unsigned short*)(ws + take((size_t)BSZ * NHID * 2));
  unsigned short* H16    = (unsigned short*)(ws + take((size_t)BSZ * NHID * 2));
  unsigned short* INT16  = (unsigned short*)(ws + take((size_t)BSZ * 2048 * 2));
  unsigned short* FIN16  = (unsigned short*)(ws + take((size_t)BSZ * 1536 * 2));
  unsigned short* HID16  = (unsigned short*)(ws + take((size_t)SEQLEN * BSZ * NHID * 2));
  float* QLIN   = (float*)(ws + take((size_t)BSZ * NHID * 4));
  float* QP     = (float*)(ws + take((size_t)BSZ * NHID * 4));
  float* INTRAW = (float*)(ws + take((size_t)BSZ * 2048 * 4));
  float* FINRAW = (float*)(ws + take((size_t)BSZ * 1536 * 4));
  float* KP     = (float*)(ws + take((size_t)BSZ * TCACHE * NHID * 4));
  float* VP     = (float*)(ws + take((size_t)BSZ * TCACHE * NHID * 4));
  int*   BARS   = (int*)  (ws + take((size_t)2048 * 4));

  // weight conversion (fp32 -> bf16, resident in L2 thereafter)
  cvt_bf16_kernel<<<512,  256, 0, stream>>>(q_w,   Wq16,   (long)512 * 1024,  (long)512 * 1024);
  cvt_bf16_kernel<<<512,  256, 0, stream>>>(ip_w,  Wip16,  (long)1536 * 512,  (long)1536 * 512);
  cvt_bf16_kernel<<<512,  256, 0, stream>>>(out_w, Wout16, (long)512 * 512,   (long)512 * 512);
  cvt_bf16_kernel<<<2048, 256, 0, stream>>>(int_w, Wint16, (long)2048 * 2048, (long)2048 * 2048);
  cvt_bf16_kernel<<<2048, 256, 0, stream>>>(fin_w, Wfin16, (long)1536 * 2048, (long)1536 * 2048);
  cvt_bf16_kernel<<<4096, 256, 0, stream>>>(dec_w, Wdec16, (long)NDECP * 512, (long)NTOK * 512);
  emb_kernel<<<1024, 256, 0, stream>>>(obs, embt, E16);
  init_kernel<<<32, 256, 0, stream>>>(BARS, H16, KP, VP, ip_b);

  RecParams P;
  P.Wq_w = Wq16; P.Wip = Wip16; P.Wout = Wout16; P.Wint = Wint16; P.Wfin = Wfin16; P.E16 = E16;
  P.Q16 = Q16; P.ATNR16 = ATNR16; P.ATNO16 = ATNO16; P.H16 = H16;
  P.INT16 = INT16; P.FIN16 = FIN16; P.HID16 = HID16;
  P.QLIN = QLIN; P.QP = QP; P.INTRAW = INTRAW; P.FINRAW = FINRAW; P.KP = KP; P.VP = VP;
  P.bars = BARS;
  P.q_b = q_b; P.qn_g = qn_g; P.qn_b = qn_b; P.ipb = ip_b; P.out_b = out_b;
  P.int_b = int_b; P.intn_g = intn_g; P.intn_b = intn_b;
  P.fin_b = fin_b; P.fn_g = fn_g; P.fn_b = fn_b;
  recurrent_kernel<<<NWG, 256, 0, stream>>>(P);

  decoder_kernel<<<dim3((NDECP / 16 + 7) / 8, SEQLEN * BSZ / 16), 256, 0, stream>>>(
      HID16, Wdec16, dec_b, (float*)d_out);
}
